// NONLocalBlock2D_68470368633115
// MI455X (gfx1250) — matrix operand order repaired and checked by reading, not compiled, not run on hardware
//
#include <hip/hip_runtime.h>
#include <hip/hip_bf16.h>
#include <stdint.h>

// ---------------------------------------------------------------------------
// NonLocalBlock2D for MI455X (gfx1250): flash-attention fused, all GEMMs on
// v_wmma_f32_16x16x32_bf16; attention K/V chunks staged to LDS with
// double-buffered global_load_async_to_lds_b128 (ASYNCcnt pipeline).
// B=4, C=128, H=W=128, Ci=64, Nq=16384, Nkv=4096.
// ---------------------------------------------------------------------------

typedef __attribute__((ext_vector_type(16))) __bf16        v16bf;
typedef __attribute__((ext_vector_type(8)))  float         v8f;
typedef __attribute__((ext_vector_type(4)))  float         v4f;
typedef __attribute__((ext_vector_type(4)))  unsigned int  v4u;

#define B_    4
#define C_    128
#define CI_   64
#define HW_   16384
#define W_IMG 128
#define NKV_  4096
#define KVC   64    // kv chunk length in attention inner loop

union V16 {            // one WMMA 16-bit operand (16 bf16 across 8 VGPRs)
  v16bf        v;
  v4u          q[2];
  unsigned int u[8];
};

__device__ __forceinline__ unsigned short f2bf(float f) {
  unsigned int u = __float_as_uint(f);
  u += 0x7FFFu + ((u >> 16) & 1u);          // round-to-nearest-even
  return (unsigned short)(u >> 16);
}
__device__ __forceinline__ float bf2f(unsigned short h) {
  return __uint_as_float(((unsigned int)h) << 16);
}
__device__ __forceinline__ unsigned int packbf(float a, float b) {
  return (unsigned int)f2bf(a) | ((unsigned int)f2bf(b) << 16);
}

// ---------------------------------------------------------------------------
// Kernel 1: 1x1 conv (GEMM): out[b, px, ci] = sum_c w[ci,c]*x[b,c,px] + bias
// Output is pixel-major bf16 [B*HW, CI] => directly a WMMA B-operand later.
// Block: 128 thr (4 waves), 64 pixels; each wave owns 16 pixels.
// ---------------------------------------------------------------------------
__global__ __launch_bounds__(128) void proj1x1_kernel(
    const float* __restrict__ x,       // [B, C, HW]
    const float* __restrict__ w,       // [CI, C]
    const float* __restrict__ bias,    // [CI]
    unsigned short* __restrict__ out)  // [B*HW, CI] bf16
{
  __shared__ __attribute__((aligned(16))) unsigned short lds_x[64 * C_];   // [px][c]
  __shared__ __attribute__((aligned(16))) unsigned short lds_w[CI_ * C_];  // [ci][c]
  const int tid   = threadIdx.x;
  const int batch = blockIdx.y;
  const int p0    = blockIdx.x * 64;

  // Stage weights [ci][c] as bf16 (coalesced: tid spans one full c-row).
  for (int ci = 0; ci < CI_; ++ci)
    lds_w[ci * C_ + tid] = f2bf(w[ci * C_ + tid]);
  // Stage x tile transposed to [px][c] (coalesced global reads over px).
  {
    const int px = tid & 63, chalf = tid >> 6;
    for (int i = 0; i < 64; ++i) {
      const int c = 2 * i + chalf;
      lds_x[px * C_ + c] = f2bf(x[(size_t)(batch * C_ + c) * HW_ + p0 + px]);
    }
  }
  __syncthreads();

  const int lane = tid & 31, wave = tid >> 5;
  const int l16 = lane & 15, h = lane >> 4;
  const int px = wave * 16 + l16;

  // B-operand fragments: lane holds n=px, K=c contiguous 16 per K-step.
  V16 xb[4];
#pragma unroll
  for (int ks = 0; ks < 4; ++ks) {
    const v4u* s = (const v4u*)&lds_x[px * C_ + ks * 32 + h * 16];
    xb[ks].q[0] = s[0]; xb[ks].q[1] = s[1];
  }

  const size_t q = (size_t)batch * HW_ + p0 + px;
#pragma unroll
  for (int mt = 0; mt < 4; ++mt) {        // 4 tiles of 16 output channels
    v8f acc = {};
#pragma unroll
    for (int ks = 0; ks < 4; ++ks) {      // K = 128 in 4 steps of 32
      V16 a;                              // A-operand: m=ci rows of w
      const unsigned short* wr = &lds_w[(mt * 16 + l16) * C_ + ks * 32 + h * 16];
      a.q[0] = *(const v4u*)wr;
      a.q[1] = *(const v4u*)(wr + 8);
      acc = __builtin_amdgcn_wmma_f32_16x16x32_bf16(false, a.v, false, xb[ks].v,
                                                    (short)0, acc, false, false);
    }
    // bias: D lane holds 8 contiguous channels (m = j + 8h)
    float bb[8];
    *(v4f*)&bb[0] = *(const v4f*)(bias + mt * 16 + 8 * h);
    *(v4f*)&bb[4] = *(const v4f*)(bias + mt * 16 + 8 * h + 4);
    float r[8];
#pragma unroll
    for (int j = 0; j < 8; ++j) r[j] = acc[j] + bb[j];
    v4u o;
    o[0] = packbf(r[0], r[1]); o[1] = packbf(r[2], r[3]);
    o[2] = packbf(r[4], r[5]); o[3] = packbf(r[6], r[7]);
    *(v4u*)&out[q * CI_ + mt * 16 + 8 * h] = o;
  }
}

// ---------------------------------------------------------------------------
// 2x2 max-pool kernels. phi -> [B*NKV, CI] (kv-major, A-operand rows for QK^T)
//                       g   -> [B, CI, NKV] (c-major,  A-operand rows for PV)
// ---------------------------------------------------------------------------
__device__ __forceinline__ float pool4(const unsigned short* in, int b, int kv, int c) {
  const int hp = kv >> 6, wp = kv & 63;
  const int p00 = (hp * 2) * W_IMG + wp * 2;
  const size_t base = (size_t)b * HW_ * CI_;
  float m =        bf2f(in[base + (size_t)(p00)             * CI_ + c]);
  m = fmaxf(m,     bf2f(in[base + (size_t)(p00 + 1)         * CI_ + c]));
  m = fmaxf(m,     bf2f(in[base + (size_t)(p00 + W_IMG)     * CI_ + c]));
  return fmaxf(m,  bf2f(in[base + (size_t)(p00 + W_IMG + 1) * CI_ + c]));
}

__global__ __launch_bounds__(256) void pool_phi_kernel(
    const unsigned short* __restrict__ in, unsigned short* __restrict__ out) {
  const int idx = blockIdx.x * 256 + threadIdx.x;  // [b][kv][c]
  const int c = idx & 63, kv = (idx >> 6) & (NKV_ - 1), b = idx >> 18;
  out[idx] = f2bf(pool4(in, b, kv, c));
}

__global__ __launch_bounds__(256) void pool_gT_kernel(
    const unsigned short* __restrict__ in, unsigned short* __restrict__ out) {
  const int idx = blockIdx.x * 256 + threadIdx.x;  // [b][c][kv]
  const int kv = idx & (NKV_ - 1), c = (idx >> 12) & 63, b = idx >> 18;
  out[idx] = f2bf(pool4(in, b, kv, c));
}

// ---------------------------------------------------------------------------
// Async staging: memory -> LDS without VGPR round trip (ASYNCcnt).
// Each wave issues 8 x global_load_async_to_lds_b128 per 16KB chunk.
// ---------------------------------------------------------------------------
__device__ __forceinline__ void async_copy_chunk(
    unsigned short* lphi, unsigned short* lg,
    const unsigned short* phiBase,   // [NKV, CI] for this batch
    const unsigned short* gBase,     // [CI, NKV] for this batch
    int kc, int tid)
{
#pragma unroll
  for (int i = 0; i < 4; ++i) {      // phi chunk: contiguous 8KB
    const int e = i * 128 + tid;     // v4u index within chunk
    const unsigned dst = (unsigned)(uintptr_t)(lphi) + (unsigned)e * 16u;
    const unsigned src = (unsigned)(kc * 512 + e) * 16u;
    asm volatile("global_load_async_to_lds_b128 %0, %1, %2"
                 :: "v"(dst), "v"(src), "s"(phiBase) : "memory");
  }
#pragma unroll
  for (int i = 0; i < 4; ++i) {      // g^T chunk: 64 rows x 128B
    const int e = i * 128 + tid, c = e >> 3, off = e & 7;
    const unsigned dst = (unsigned)(uintptr_t)(lg) + (unsigned)e * 16u;
    const unsigned src = (unsigned)(c * (NKV_ / 8) + kc * (KVC / 8) + off) * 16u;
    asm volatile("global_load_async_to_lds_b128 %0, %1, %2"
                 :: "v"(dst), "v"(src), "s"(gBase) : "memory");
  }
}

// ---------------------------------------------------------------------------
// Kernel 3: fused flash attention.  Per wave: 16 queries, loop kv in chunks
// of 64.  S^T = phi^T x theta^T  (q in lanes -> per-lane online softmax),
// y^T += g^T x P^T.  4 waves share double-buffered LDS chunks filled by the
// async-DMA path; next chunk copies overlap current chunk's WMMAs.
// ---------------------------------------------------------------------------
__global__ __launch_bounds__(128) void attn_kernel(
    const unsigned short* __restrict__ theta,  // [B*HW, CI]
    const unsigned short* __restrict__ phi,    // [B*NKV, CI]
    const unsigned short* __restrict__ gT,     // [B, CI, NKV]
    unsigned short* __restrict__ y)            // [B*HW, CI]
{
  __shared__ __attribute__((aligned(16))) unsigned short lds_phi[2][KVC * CI_]; // [kv][c]
  __shared__ __attribute__((aligned(16))) unsigned short lds_g[2][CI_ * KVC];   // [c][kv]
  const int tid = threadIdx.x;
  const int batch = blockIdx.y;
  const int qb = blockIdx.x * 64;
  const int lane = tid & 31, wave = tid >> 5;
  const int l16 = lane & 15, h = lane >> 4;
  const bool lo = (h == 0);
  const int qpix = qb + wave * 16 + l16;
  const size_t qrow = ((size_t)batch * HW_ + qpix) * CI_;

  // theta B-operand fragments (kept in registers for whole kv loop)
  V16 tb[2];
  {
    const v4u* t0 = (const v4u*)&theta[qrow + h * 16];
    tb[0].q[0] = t0[0]; tb[0].q[1] = t0[1];
    const v4u* t1 = (const v4u*)&theta[qrow + 32 + h * 16];
    tb[1].q[0] = t1[0]; tb[1].q[1] = t1[1];
  }

  float m_run = -1e30f, l_run = 0.f;
  v8f zero = {};
  v8f yacc[4]; yacc[0] = zero; yacc[1] = zero; yacc[2] = zero; yacc[3] = zero;

  const unsigned short* phiB = &phi[(size_t)batch * NKV_ * CI_];
  const unsigned short* gB   = &gT[(size_t)batch * CI_ * NKV_];
  const int NCH = NKV_ / KVC;

  // Prime the pipeline with chunk 0.
  async_copy_chunk(lds_phi[0], lds_g[0], phiB, gB, 0, tid);

  for (int kc = 0; kc < NCH; ++kc) {
    const int cur = kc & 1;
    if (kc + 1 < NCH) {
      // Kick off next chunk into the other buffer, then wait only for the
      // current chunk (8 newest async ops may remain outstanding).
      async_copy_chunk(lds_phi[cur ^ 1], lds_g[cur ^ 1], phiB, gB, kc + 1, tid);
      asm volatile("s_wait_asynccnt 0x8" ::: "memory");
    } else {
      asm volatile("s_wait_asynccnt 0x0" ::: "memory");
    }
    __syncthreads();   // all waves' chunk-kc data visible in LDS

    const unsigned short* Lphi = lds_phi[cur];
    const unsigned short* Lg   = lds_g[cur];

    // ---- scores: 4 S^T tiles of 16kv x 16q, K = Ci = 64 ----
    v8f st[4];
#pragma unroll
    for (int t = 0; t < 4; ++t) {
      v8f acc = {};
#pragma unroll
      for (int ks = 0; ks < 2; ++ks) {
        V16 a;  // A-operand: m = kv rows of phi
        const unsigned short* pr = &Lphi[(t * 16 + l16) * CI_ + ks * 32 + h * 16];
        a.q[0] = *(const v4u*)pr;
        a.q[1] = *(const v4u*)(pr + 8);
        acc = __builtin_amdgcn_wmma_f32_16x16x32_bf16(false, a.v, false, tb[ks].v,
                                                      (short)0, acc, false, false);
      }
      st[t] = acc;
    }

    // ---- online softmax (q lives in lane; halves combined via shfl_xor 16) ----
    float mloc = -1e30f;
#pragma unroll
    for (int t = 0; t < 4; ++t)
#pragma unroll
      for (int j = 0; j < 8; ++j) mloc = fmaxf(mloc, st[t][j]);
    mloc = fmaxf(mloc, __shfl_xor(mloc, 16, 32));
    const float m_new = fmaxf(m_run, mloc);
    const float corr = __expf(m_run - m_new);
    float lsum = 0.f;
    unsigned int pp[4][4];
#pragma unroll
    for (int t = 0; t < 4; ++t) {
      float e[8];
#pragma unroll
      for (int j = 0; j < 8; ++j) { e[j] = __expf(st[t][j] - m_new); lsum += e[j]; }
#pragma unroll
      for (int p = 0; p < 4; ++p) pp[t][p] = packbf(e[2 * p], e[2 * p + 1]);
    }
    l_run = l_run * corr + lsum + __shfl_xor(lsum, 16, 32);
    m_run = m_new;

    // D-layout -> B-layout for P^T: exchange lane-halves, select per half.
    unsigned int xp[4][4];
#pragma unroll
    for (int t = 0; t < 4; ++t)
#pragma unroll
      for (int p = 0; p < 4; ++p) xp[t][p] = __shfl_xor(pp[t][p], 16, 32);

    V16 pb0, pb1;
#pragma unroll
    for (int j = 0; j < 4; ++j) {
      pb0.u[j]     = lo ? pp[0][j] : xp[1][j];
      pb0.u[j + 4] = lo ? xp[0][j] : pp[1][j];
      pb1.u[j]     = lo ? pp[2][j] : xp[3][j];
      pb1.u[j + 4] = lo ? xp[2][j] : pp[3][j];
    }

    // ---- y^T += g^T x P^T over 64 kv (2 K-steps), 4 c-tiles ----
#pragma unroll
    for (int ct = 0; ct < 4; ++ct) {
      v8f acc = yacc[ct];
#pragma unroll
      for (int j = 0; j < 8; ++j) acc[j] *= corr;   // rescale running output
#pragma unroll
      for (int ks = 0; ks < 2; ++ks) {
        V16 a;  // A-operand: m = c rows of g^T
        const unsigned short* gr = &Lg[(ct * 16 + l16) * KVC + ks * 32 + h * 16];
        a.q[0] = *(const v4u*)gr;
        a.q[1] = *(const v4u*)(gr + 8);
        acc = __builtin_amdgcn_wmma_f32_16x16x32_bf16(false, a.v, false,
                                                      ks == 0 ? pb0.v : pb1.v,
                                                      (short)0, acc, false, false);
      }
      yacc[ct] = acc;
    }
    __syncthreads();   // all waves done reading buffer `cur` before its reuse
  }

  const float inv = 1.f / l_run;
#pragma unroll
  for (int ct = 0; ct < 4; ++ct) {
    float r[8];
#pragma unroll
    for (int j = 0; j < 8; ++j) r[j] = yacc[ct][j] * inv;
    v4u o;
    o[0] = packbf(r[0], r[1]); o[1] = packbf(r[2], r[3]);
    o[2] = packbf(r[4], r[5]); o[3] = packbf(r[6], r[7]);
    *(v4u*)&y[qrow + ct * 16 + 8 * h] = o;   // y stored q-major [B*HW, CI]
  }
}

// ---------------------------------------------------------------------------
// Kernel 4: out[b,co,px] = W_w[co,:] . y[px,:] + W_b[co] + x[b,co,px]
// ---------------------------------------------------------------------------
__global__ __launch_bounds__(128) void wconv_kernel(
    const unsigned short* __restrict__ y,  // [B*HW, CI]
    const float* __restrict__ ww,          // [C, CI]
    const float* __restrict__ wb,          // [C]
    const float* __restrict__ x,           // [B, C, HW]
    float* __restrict__ out)               // [B, C, HW]
{
  __shared__ __attribute__((aligned(16))) unsigned short lds_w[C_ * CI_]; // [co][ci]
  const int tid = threadIdx.x;
  const int batch = blockIdx.y;
  const int p0 = blockIdx.x * 64;
  {
    const int ci = tid & 63, chalf = tid >> 6;
    for (int i = 0; i < 64; ++i) {
      const int co = 2 * i + chalf;
      lds_w[co * CI_ + ci] = f2bf(ww[co * CI_ + ci]);
    }
  }
  __syncthreads();

  const int lane = tid & 31, wave = tid >> 5;
  const int l16 = lane & 15, h = lane >> 4;
  const int px = p0 + wave * 16 + l16;
  const size_t qrow = ((size_t)batch * HW_ + px) * CI_;

  V16 yb[2];
  {
    const v4u* y0 = (const v4u*)&y[qrow + h * 16];
    yb[0].q[0] = y0[0]; yb[0].q[1] = y0[1];
    const v4u* y1 = (const v4u*)&y[qrow + 32 + h * 16];
    yb[1].q[0] = y1[0]; yb[1].q[1] = y1[1];
  }

#pragma unroll
  for (int mt = 0; mt < 8; ++mt) {   // 128 output channels in 8 tiles
    v8f acc = {};
#pragma unroll
    for (int ks = 0; ks < 2; ++ks) {
      V16 a;
      const unsigned short* wr = &lds_w[(mt * 16 + l16) * CI_ + ks * 32 + h * 16];
      a.q[0] = *(const v4u*)wr;
      a.q[1] = *(const v4u*)(wr + 8);
      acc = __builtin_amdgcn_wmma_f32_16x16x32_bf16(false, a.v, false, yb[ks].v,
                                                    (short)0, acc, false, false);
    }
#pragma unroll
    for (int j = 0; j < 8; ++j) {
      const int co = mt * 16 + 8 * h + j;
      const size_t oi = ((size_t)batch * C_ + co) * HW_ + px;
      out[oi] = acc[j] + wb[co] + x[oi];
    }
  }
}

// ---------------------------------------------------------------------------
extern "C" void kernel_launch(void* const* d_in, const int* in_sizes, int n_in,
                              void* d_out, int out_size, void* d_ws, size_t ws_size,
                              hipStream_t stream) {
  (void)in_sizes; (void)n_in; (void)out_size; (void)ws_size;
  const float* x       = (const float*)d_in[0];
  const float* theta_w = (const float*)d_in[1];
  const float* theta_b = (const float*)d_in[2];
  const float* phi_w   = (const float*)d_in[3];
  const float* phi_b   = (const float*)d_in[4];
  const float* g_w     = (const float*)d_in[5];
  const float* g_b     = (const float*)d_in[6];
  const float* W_w     = (const float*)d_in[7];
  const float* W_b     = (const float*)d_in[8];
  float* out = (float*)d_out;

  char* ws = (char*)d_ws;                                   // ~36 MB used
  unsigned short* thetaT = (unsigned short*)(ws + 0);        // 8 MB [B*HW,CI]
  unsigned short* phiF   = (unsigned short*)(ws + 8388608);  // 8 MB full-res
  unsigned short* gF     = (unsigned short*)(ws + 16777216); // 8 MB full-res
  unsigned short* phiP   = (unsigned short*)(ws + 25165824); // 2 MB [B*NKV,CI]
  unsigned short* gT     = (unsigned short*)(ws + 27262976); // 2 MB [B,CI,NKV]
  unsigned short* yv     = (unsigned short*)(ws + 29360128); // 8 MB [B*HW,CI]

  const dim3 gproj(HW_ / 64, B_);
  proj1x1_kernel<<<gproj, 128, 0, stream>>>(x, theta_w, theta_b, thetaT);
  proj1x1_kernel<<<gproj, 128, 0, stream>>>(x, phi_w,   phi_b,   phiF);
  proj1x1_kernel<<<gproj, 128, 0, stream>>>(x, g_w,     g_b,     gF);
  pool_phi_kernel<<<(B_ * NKV_ * CI_) / 256, 256, 0, stream>>>(phiF, phiP);
  pool_gT_kernel<<<(B_ * NKV_ * CI_) / 256, 256, 0, stream>>>(gF, gT);
  attn_kernel<<<dim3(HW_ / 64, B_), 128, 0, stream>>>(thetaT, phiP, gT, yv);
  wconv_kernel<<<dim3(HW_ / 64, B_), 128, 0, stream>>>(yv, W_w, W_b, x, out);
}
